// AttentionModule_87686052315171
// MI455X (gfx1250) — matrix operand order repaired and checked by reading, not compiled, not run on hardware
//
#include <hip/hip_runtime.h>
#include <hip/hip_bf16.h>

typedef __attribute__((ext_vector_type(16))) __bf16 v16bf;
typedef __attribute__((ext_vector_type(8)))  float  v8f;

#define R_DIM 512
#define M_DIM 256
#define A_DIM 256
#define B_DIM 64
#define L_DIM 4096
#define NCHUNK 8          // L split across workgroups per batch
#define NWAVE 8           // waves per workgroup
#define LPAIRS 2          // each wave: 2 iterations x 32 rows = 64 rows
#define PART_STRIDE (M_DIM + 2)

// ---------- prep: Ua f32 -> bf16 (row-major [A][M]) ----------
__global__ void prep_ua_kernel(const float* __restrict__ Ua,
                               __bf16* __restrict__ UaB) {
  int i = blockIdx.x * blockDim.x + threadIdx.x;
  UaB[i] = (__bf16)Ua[i];
}

// ---------- prep: q[b][a] = hidden[b] . Wa[a] + ba[a] ----------
__global__ void prep_q_kernel(const float* __restrict__ hidden,
                              const float* __restrict__ Wa,
                              const float* __restrict__ ba,
                              float* __restrict__ qout) {
  const int b = blockIdx.x;
  const int a = threadIdx.x;
  const float* h  = hidden + b * R_DIM;
  const float* wr = Wa + a * R_DIM;
  float acc = ba[a];
  #pragma unroll 4
  for (int r = 0; r < R_DIM; r += 4) {
    acc += h[r] * wr[r] + h[r + 1] * wr[r + 1] +
           h[r + 2] * wr[r + 2] + h[r + 3] * wr[r + 3];
  }
  qout[b * A_DIM + a] = acc;
}

__device__ __forceinline__ float fast_tanh(float x) {
#if __has_builtin(__builtin_amdgcn_tanhf)
  return __builtin_amdgcn_tanhf(x);          // native v_tanh_f32 (CDNA5 trans op)
#elif __has_builtin(__builtin_amdgcn_tanh_f32)
  return __builtin_amdgcn_tanh_f32(x);
#else
  float e2 = __expf(2.0f * x);               // fallback: 1 - 2/(exp(2x)+1)
  return 1.0f - 2.0f / (e2 + 1.0f);
#endif
}

// ---------- main: fused k-GEMM (bf16 WMMA) + tanh·w + online softmax + ctx ----------
__launch_bounds__(256)
__global__ void attn_main_kernel(const float* __restrict__ fv,
                                 const __bf16* __restrict__ UaB,
                                 const float* __restrict__ qbuf,
                                 const float* __restrict__ wvec,
                                 float* __restrict__ partials) {
  const int chunk = blockIdx.x;          // 0..NCHUNK-1
  const int b     = blockIdx.y;          // 0..B-1
  const int tid   = threadIdx.x;
  const int wid   = tid >> 5;            // wave in workgroup (wave32)
  const int lane  = tid & 31;
  const int half  = lane >> 4;           // 0: lanes 0-15, 1: lanes 16-31
  const int r16   = lane & 15;

  __shared__ float lds_m[NWAVE];
  __shared__ float lds_s[NWAVE];
  __shared__ float lds_ctx[NWAVE][M_DIM];

  const float* fvb   = fv + (size_t)b * L_DIM * M_DIM;
  const int    lbase = chunk * (L_DIM / NCHUNK) + wid * (LPAIRS * 32);

  float m_run = -3.0e38f;
  float s_run = 0.0f;
  float ctx[8];
  #pragma unroll
  for (int j = 0; j < 8; ++j) ctx[j] = 0.0f;

  for (int lp = 0; lp < LPAIRS; ++lp) {
    const int l0 = lbase + lp * 32;      // this iteration covers rows l0..l0+31

    // ---- A fragments for TWO 16-row tiles: fv f32 -> bf16, 16x16x32 A-layout ----
    // lane(half,r16): row base+r16, K = m0+half*16..+15 (elems 0-15, contiguous)
    v16bf afr0[8], afr1[8];
    {
      const float* rp0 = fvb + (size_t)(l0 + r16) * M_DIM;
      const float* rp1 = fvb + (size_t)(l0 + 16 + r16) * M_DIM;
      #pragma unroll
      for (int ks = 0; ks < 8; ++ks) {
        const float* p0 = rp0 + ks * 32 + half * 16;
        const float* p1 = rp1 + ks * 32 + half * 16;
        v16bf a0, a1;
        #pragma unroll
        for (int i = 0; i < 8; ++i) {
          a0[i]     = (__bf16)p0[i];
          a0[i + 8] = (__bf16)p0[i + 8];
          a1[i]     = (__bf16)p1[i];
          a1[i + 8] = (__bf16)p1[i + 8];
        }
        afr0[ks] = a0;
        afr1[ks] = a1;
      }
    }
    if (lp + 1 < LPAIRS)  // gfx1250 global_prefetch_b8 for next 32-row block
      __builtin_prefetch(fvb + (size_t)(l0 + 32 + r16) * M_DIM + half * 128, 0, 1);

    float esum0[8], esum1[8];
    #pragma unroll
    for (int r = 0; r < 8; ++r) { esum0[r] = 0.0f; esum1[r] = 0.0f; }

    // ---- A (output-feature) tiles: each B fragment feeds BOTH row tiles ----
    for (int at = 0; at < 16; ++at) {
      const int a0 = at * 16;
      const float qv = qbuf[b * A_DIM + a0 + r16];
      const float wv = wvec[a0 + r16];
      // B fragment: lane holds Ua[a0+r16][m0 + half*16 .. +15], contiguous bf16
      const __bf16* bp = UaB + (a0 + r16) * M_DIM + half * 16;

      v8f c0 = {}, c1 = {};
      #pragma unroll
      for (int ks = 0; ks < 8; ++ks) {
        v16bf bfr = *(const v16bf*)(bp + ks * 32);
        c0 = __builtin_amdgcn_wmma_f32_16x16x32_bf16(
                 false, afr0[ks], false, bfr, (short)0, c0, false, false);
        c1 = __builtin_amdgcn_wmma_f32_16x16x32_bf16(
                 false, afr1[ks], false, bfr, (short)0, c1, false, false);
      }
      // C layout: element r -> row (r + 8*half), col (a0 + r16)
      #pragma unroll
      for (int r = 0; r < 8; ++r) {
        esum0[r] += fast_tanh(c0[r] + qv) * wv;
        esum1[r] += fast_tanh(c1[r] + qv) * wv;
      }
    }

    // ---- reduce e over the 16 lanes of each half (butterfly) ----
    #pragma unroll
    for (int off = 1; off < 16; off <<= 1) {
      #pragma unroll
      for (int r = 0; r < 8; ++r) {
        esum0[r] += __shfl_xor(esum0[r], off, 32);
        esum1[r] += __shfl_xor(esum1[r], off, 32);
      }
    }
    // lanes 0-15: e[rows base+0..7]; lanes 16-31: e[rows base+8..15] (per tile)

    // ---- online softmax update over 32 rows ----
    float mx = fmaxf(esum0[0], esum1[0]);
    #pragma unroll
    for (int r = 1; r < 8; ++r) mx = fmaxf(mx, fmaxf(esum0[r], esum1[r]));
    mx = fmaxf(mx, __shfl_xor(mx, 16, 32));
    const float new_m = fmaxf(m_run, mx);
    const float scale = __expf(m_run - new_m);

    float p0[8], p1[8];
    float psum = 0.0f;
    #pragma unroll
    for (int r = 0; r < 8; ++r) {
      p0[r] = __expf(esum0[r] - new_m);
      p1[r] = __expf(esum1[r] - new_m);
      psum += p0[r] + p1[r];
    }
    psum += __shfl_xor(psum, 16, 32);
    s_run = s_run * scale + psum;
    m_run = new_m;

    #pragma unroll
    for (int j = 0; j < 8; ++j) ctx[j] *= scale;

    // ---- ctx += p * fv rows (re-read hits L0/L2; coalesced, lane -> m) ----
    #pragma unroll
    for (int r = 0; r < 8; ++r) {
      const float pm0 = p0[r], po0 = __shfl_xor(p0[r], 16, 32);
      const float pm1 = p1[r], po1 = __shfl_xor(p1[r], 16, 32);
      const float* rm0 = fvb + (size_t)(l0 + half * 8 + r) * M_DIM;
      const float* ro0 = fvb + (size_t)(l0 + (1 - half) * 8 + r) * M_DIM;
      const float* rm1 = fvb + (size_t)(l0 + 16 + half * 8 + r) * M_DIM;
      const float* ro1 = fvb + (size_t)(l0 + 16 + (1 - half) * 8 + r) * M_DIM;
      #pragma unroll
      for (int j = 0; j < 8; ++j) {
        const int mi = lane + 32 * j;
        ctx[j] += pm0 * rm0[mi] + po0 * ro0[mi] + pm1 * rm1[mi] + po1 * ro1[mi];
      }
    }
  }

  // ---- intra-workgroup combine of partial softmax states ----
  if (lane == 0) { lds_m[wid] = m_run; lds_s[wid] = s_run; }
  #pragma unroll
  for (int j = 0; j < 8; ++j) lds_ctx[wid][lane + 32 * j] = ctx[j];
  __syncthreads();

  float Mx = -3.0e38f;
  #pragma unroll
  for (int w2 = 0; w2 < NWAVE; ++w2) Mx = fmaxf(Mx, lds_m[w2]);
  float S = 0.0f, cf = 0.0f;
  #pragma unroll
  for (int w2 = 0; w2 < NWAVE; ++w2) {
    const float sc = __expf(lds_m[w2] - Mx);
    S  += lds_s[w2] * sc;
    cf += lds_ctx[w2][tid] * sc;
  }
  float* part = partials + (size_t)(b * NCHUNK + chunk) * PART_STRIDE;
  if (tid == 0) { part[0] = Mx; part[1] = S; }
  part[2 + tid] = cf;
}

// ---------- final cross-chunk softmax reduction ----------
__global__ void attn_reduce_kernel(const float* __restrict__ partials,
                                   float* __restrict__ out) {
  const int b = blockIdx.x;
  const int t = threadIdx.x;
  const float* base = partials + (size_t)b * NCHUNK * PART_STRIDE;
  float Mx = -3.0e38f;
  #pragma unroll
  for (int c = 0; c < NCHUNK; ++c) Mx = fmaxf(Mx, base[c * PART_STRIDE]);
  float S = 0.0f, cf = 0.0f;
  #pragma unroll
  for (int c = 0; c < NCHUNK; ++c) {
    const float* p = base + c * PART_STRIDE;
    const float sc = __expf(p[0] - Mx);
    S  += p[1] * sc;
    cf += p[2 + t] * sc;
  }
  out[b * M_DIM + t] = cf / S;
}

extern "C" void kernel_launch(void* const* d_in, const int* in_sizes, int n_in,
                              void* d_out, int out_size, void* d_ws, size_t ws_size,
                              hipStream_t stream) {
  const float* hidden = (const float*)d_in[0];   // [B, R]
  const float* fv     = (const float*)d_in[1];   // [B, L, M]
  const float* Wa     = (const float*)d_in[2];   // [A, R]
  const float* Ua     = (const float*)d_in[3];   // [A, M]
  const float* w      = (const float*)d_in[4];   // [A, 1]
  const float* ba     = (const float*)d_in[5];   // [1, A]
  float* out = (float*)d_out;

  char* ws = (char*)d_ws;
  __bf16* UaB     = (__bf16*)ws;                              // 128 KB
  float*  qbuf    = (float*)(ws + 131072);                    // 64 KB
  float*  partial = (float*)(ws + 131072 + 65536);            // B*NCHUNK*258*4 B

  prep_ua_kernel<<<dim3((A_DIM * M_DIM) / 256), 256, 0, stream>>>(Ua, UaB);
  prep_q_kernel<<<dim3(B_DIM), A_DIM, 0, stream>>>(hidden, Wa, ba, qbuf);
  attn_main_kernel<<<dim3(NCHUNK, B_DIM), NWAVE * 32, 0, stream>>>(fv, UaB, qbuf, w, partial);
  attn_reduce_kernel<<<dim3(B_DIM), M_DIM, 0, stream>>>(partial, out);
}
